// LingoAuraLLM_91207925498283
// MI455X (gfx1250) — matrix operand order repaired and checked by reading, not compiled, not run on hardware
//
#include <hip/hip_runtime.h>
#include <hip/hip_bf16.h>
#include <math.h>

// ---------------------------------------------------------------------------
// d_in leaf order (jax tree flatten: dicts sorted by key, lists/tuples in order)
//  0 input_ids(B,T)i32  1 attention_mask(B,T)f32  2 labels(B,T)i32
//  3 visual(B,256,1024) 4 acoustic(B,512,768)
//  5..10  ap: b,b1,b2,g,w1(1536x768),w2(2048x1536)
//  11 embed(32000x2048) 12 final_norm(2048)
//  13+17*l per layer: attn_norm, down.scales, gate.scales, k.A,k.B,k.scales,
//                     mlp_norm, o.A,o.B,o.scales, q.A,q.B,q.scales,
//                     up.scales, v.A,v.B,v.scales
//  47 lm_head(32000x2048) 48 temp
//  49..54 vp: b,b1,b2,g,w1(2048x1024),w2(2048x2048)
//  55+7*l qcodes per layer: down,gate,k,o,q,up,v   (int32, N x K)
// ---------------------------------------------------------------------------

typedef __attribute__((ext_vector_type(16))) _Float16 v16h;
typedef __attribute__((ext_vector_type(8)))  float    v8f;

#define H_DIM   2048
#define SEQ     514
#define T_TOK   512
#define NHEAD   16
#define HDIM    128
#define FFN_DIM 5632
#define VOCAB   32000
#define RANK    32
#define BATCH   2
#define MROWS   (BATCH*SEQ)          // 1028
#define NQT     33                   // ceil(SEQ/16)
#define NKT     34                   // padded 16-wide key tiles (SKP/16)
#define NKC     17                   // padded 32-wide key chunks
#define SKP     544                  // NKC*32
#define VC      3200                 // LM-head chunk width (multiple of 128)
#define NCHUNK  10                   // VOCAB / VC

__constant__ float NF4_LUT[16] = {
  -1.0f, -0.6961928009986877f, -0.5250730514526367f, -0.39491748809814453f,
  -0.28444138169288635f, -0.18477343022823334f, -0.09105003625154495f, 0.0f,
  0.07958029955625534f, 0.16093020141124725f, 0.2461123913526535f,
  0.33791524171829224f, 0.44070982933044434f, 0.5626170039176941f,
  0.7229568362236023f, 1.0f };

enum { GF_NF4 = 1, GF_RELU = 2, GF_ACC = 4, GF_BIAS = 8 };

union HPack8  { _Float16 h[8];  uint4 u; };
union HPack16 { _Float16 h[16]; v16h  v; };

__device__ inline v8f wmma_f16(v16h a, v16h b, v8f c) {
  return __builtin_amdgcn_wmma_f32_16x16x32_f16(false, a, false, b, (short)0, c,
                                                false, false);
}

// ===========================================================================
// Generic WMMA GEMM:  Y[M,N] = alpha * X[M,K] @ W[N,K]^T (+bias)(relu)(+=Y)
// W source: f32 rows (Wf) or NF4 codes (Wq int32 + scales N x K/64).
// Block tile 128x128, 8 waves (4x2), each wave 32x64 -> 8 wmma per K-step.
// Double-buffered LDS (ping-pong) with register prefetch of the next K-step:
// one __syncthreads per step, global loads overlap the tensor pipe.
// NF4 LUT lives in LDS (16 distinct banks -> conflict-free ds_load_b32
// gathers) instead of per-lane global gathers from __constant__ memory.
// ===========================================================================
__global__ __launch_bounds__(256)
void gemm_kernel(const float* __restrict__ X,
                 const float* __restrict__ Wf,
                 const int*   __restrict__ Wq,
                 const float* __restrict__ scales,
                 const float* __restrict__ bias,
                 float*       __restrict__ Y,
                 int M, int N, int K, int flags, float alpha)
{
  __shared__ _Float16 lA[2][8][32][16];   // [buf][m-strip][lane][half]
  __shared__ _Float16 lB[2][8][32][16];   // [buf][n-strip][lane][half]
  __shared__ float    lutS[16];           // NF4 dequant table (bank-parallel)

  const int t    = threadIdx.x;
  const int lane = t & 31;
  const int w    = t >> 5;
  const int wm   = w >> 1;             // 0..3  (rows 32*wm..+31)
  const int wn   = w & 1;              // 0..1  (cols 64*wn..+63)
  const int m0   = blockIdx.y * 128;
  const int n0   = blockIdx.x * 128;

  v8f acc[2][4] = {};
  const int nkt = K >> 5;

  // staging task geometry: thread handles (row t&127, octets t>>7 and t>>7 + 2)
  const int srow   = t & 127;
  const int skoct0 = t >> 7;           // 0..1

  float4 raf[2][2];                    // A prefetch regs
  float4 rbf[2][2];                    // B prefetch regs (f32 mode)
  int4   rbq[2][2];                    // B prefetch regs (NF4 mode)
  float  rbs[2];                       // NF4 scales

  auto loadA = [&](int kt) {
    int gm = m0 + srow;
#pragma unroll
    for (int i = 0; i < 2; ++i) {
      int koct = skoct0 + 2 * i;
      if (gm < M) {
        const float4* p = (const float4*)(X + (size_t)gm * K + (kt << 5) + koct * 8);
        raf[i][0] = p[0]; raf[i][1] = p[1];
      } else {
        raf[i][0] = make_float4(0.f, 0.f, 0.f, 0.f);
        raf[i][1] = make_float4(0.f, 0.f, 0.f, 0.f);
      }
    }
  };
  auto storeA = [&](int buf) {
#pragma unroll
    for (int i = 0; i < 2; ++i) {
      int koct = skoct0 + 2 * i;
      HPack8 pk;
      pk.h[0]=(_Float16)raf[i][0].x; pk.h[1]=(_Float16)raf[i][0].y;
      pk.h[2]=(_Float16)raf[i][0].z; pk.h[3]=(_Float16)raf[i][0].w;
      pk.h[4]=(_Float16)raf[i][1].x; pk.h[5]=(_Float16)raf[i][1].y;
      pk.h[6]=(_Float16)raf[i][1].z; pk.h[7]=(_Float16)raf[i][1].w;
      int ln  = (srow & 15) + 16 * (koct & 1);
      int idx = (koct >> 1) * 8;
      *(uint4*)&lA[buf][srow >> 4][ln][idx] = pk.u;
    }
  };
  auto loadB = [&](int kt) {
    int gn = n0 + srow;
    if (gn < N) {
      if (flags & GF_NF4) {
        // deep L2 prefetch of the dominant weight-code stream
        if (kt + 4 < nkt)
          __builtin_prefetch(Wq + (size_t)gn * K + ((kt + 4) << 5) + skoct0 * 8, 0, 1);
#pragma unroll
        for (int i = 0; i < 2; ++i) {
          int koct = skoct0 + 2 * i;
          const int4* p = (const int4*)(Wq + (size_t)gn * K + (kt << 5) + koct * 8);
          rbq[i][0] = p[0]; rbq[i][1] = p[1];
          rbs[i] = scales[(size_t)gn * (K >> 6) + (((kt << 5) + koct * 8) >> 6)];
        }
      } else {
#pragma unroll
        for (int i = 0; i < 2; ++i) {
          int koct = skoct0 + 2 * i;
          const float4* p = (const float4*)(Wf + (size_t)gn * K + (kt << 5) + koct * 8);
          rbf[i][0] = p[0]; rbf[i][1] = p[1];
        }
      }
    } else {
#pragma unroll
      for (int i = 0; i < 2; ++i) {
        rbq[i][0] = make_int4(0, 0, 0, 0); rbq[i][1] = make_int4(0, 0, 0, 0);
        rbf[i][0] = make_float4(0.f, 0.f, 0.f, 0.f);
        rbf[i][1] = make_float4(0.f, 0.f, 0.f, 0.f);
        rbs[i] = 0.f;
      }
    }
  };
  auto storeB = [&](int buf) {
#pragma unroll
    for (int i = 0; i < 2; ++i) {
      int koct = skoct0 + 2 * i;
      HPack8 pk;
      if (flags & GF_NF4) {
        float s = rbs[i];
        pk.h[0]=(_Float16)(lutS[rbq[i][0].x & 15] * s);
        pk.h[1]=(_Float16)(lutS[rbq[i][0].y & 15] * s);
        pk.h[2]=(_Float16)(lutS[rbq[i][0].z & 15] * s);
        pk.h[3]=(_Float16)(lutS[rbq[i][0].w & 15] * s);
        pk.h[4]=(_Float16)(lutS[rbq[i][1].x & 15] * s);
        pk.h[5]=(_Float16)(lutS[rbq[i][1].y & 15] * s);
        pk.h[6]=(_Float16)(lutS[rbq[i][1].z & 15] * s);
        pk.h[7]=(_Float16)(lutS[rbq[i][1].w & 15] * s);
      } else {
        pk.h[0]=(_Float16)rbf[i][0].x; pk.h[1]=(_Float16)rbf[i][0].y;
        pk.h[2]=(_Float16)rbf[i][0].z; pk.h[3]=(_Float16)rbf[i][0].w;
        pk.h[4]=(_Float16)rbf[i][1].x; pk.h[5]=(_Float16)rbf[i][1].y;
        pk.h[6]=(_Float16)rbf[i][1].z; pk.h[7]=(_Float16)rbf[i][1].w;
      }
      int ln  = (srow & 15) + 16 * (koct & 1);
      int idx = (koct >> 1) * 8;
      *(uint4*)&lB[buf][srow >> 4][ln][idx] = pk.u;
    }
  };

  // prologue: publish LUT, stage K-step 0 into buffer 0
  if (t < 16) lutS[t] = NF4_LUT[t];
  loadA(0); loadB(0);
  __syncthreads();                     // LUT visible before first dequant
  storeA(0); storeB(0);
  __syncthreads();

  for (int kt = 0; kt < nkt; ++kt) {
    const int cur = kt & 1;
    const bool has = (kt + 1) < nkt;
    if (has) { loadA(kt + 1); loadB(kt + 1); }   // global prefetch overlaps wmma

    v16h a0 = *(const v16h*)&lA[cur][2 * wm + 0][lane][0];
    v16h a1 = *(const v16h*)&lA[cur][2 * wm + 1][lane][0];
    v16h b0 = *(const v16h*)&lB[cur][4 * wn + 0][lane][0];
    v16h b1 = *(const v16h*)&lB[cur][4 * wn + 1][lane][0];
    v16h b2 = *(const v16h*)&lB[cur][4 * wn + 2][lane][0];
    v16h b3 = *(const v16h*)&lB[cur][4 * wn + 3][lane][0];
    acc[0][0] = wmma_f16(a0, b0, acc[0][0]);
    acc[0][1] = wmma_f16(a0, b1, acc[0][1]);
    acc[0][2] = wmma_f16(a0, b2, acc[0][2]);
    acc[0][3] = wmma_f16(a0, b3, acc[0][3]);
    acc[1][0] = wmma_f16(a1, b0, acc[1][0]);
    acc[1][1] = wmma_f16(a1, b1, acc[1][1]);
    acc[1][2] = wmma_f16(a1, b2, acc[1][2]);
    acc[1][3] = wmma_f16(a1, b3, acc[1][3]);

    if (has) {
      // WAR on buf cur^1 was fenced by the barrier at end of previous step
      storeA(cur ^ 1); storeB(cur ^ 1);
      __syncthreads();                           // single barrier per K-step
    }
  }

  // ---- epilogue (C layout: lane = n%16 + 16*(m>=8), vgpr r = m%8) ----
  const int nn   = lane & 15;
  const int mofs = (lane >> 4) * 8;
#pragma unroll
  for (int i = 0; i < 2; ++i)
#pragma unroll
    for (int j = 0; j < 4; ++j) {
      int gn = n0 + (4 * wn + j) * 16 + nn;
      if (gn >= N) continue;
      float bv = (flags & GF_BIAS) ? bias[gn] : 0.f;
#pragma unroll
      for (int r = 0; r < 8; ++r) {
        int gm = m0 + (2 * wm + i) * 16 + mofs + r;
        if (gm >= M) continue;
        float v = alpha * acc[i][j][r] + bv;
        if (flags & GF_RELU) v = v > 0.f ? v : 0.f;
        size_t o = (size_t)gm * N + gn;
        if (flags & GF_ACC) Y[o] += v; else Y[o] = v;
      }
    }
}

// ===========================================================================
// WMMA attention: block = 4 waves, one (batch, head, 16-query strip).
// Phase1: scores via QK^T wmma (Q frags in regs, K straight to B-frag).
// Phase2: softmax in LDS, P written in A-frag f16 layout (unnormalized).
// Phase3: O = P @ V via wmma, V staged to B-frag LDS, divide by rowsum.
// ===========================================================================
__global__ __launch_bounds__(128)
void attn_kernel(const float* __restrict__ q, const float* __restrict__ k,
                 const float* __restrict__ v, const float* __restrict__ me,
                 float* __restrict__ o)
{
  __shared__ float     sc[16][SKP];
  __shared__ _Float16  pf[NKC][32][16];
  __shared__ _Float16  vf[8][32][16];
  __shared__ float     red[16][8];
  __shared__ float     rsum[16];
  __shared__ float     rmax[16];

  const int t    = threadIdx.x;
  const int lane = t & 31;
  const int w    = t >> 5;                 // 0..3
  const int bid  = blockIdx.x;
  const int qt   = bid % NQT;
  const int hd   = (bid / NQT) % NHEAD;
  const int b    = bid / (NQT * NHEAD);
  const int q0   = qt * 16;

  const int m  = lane & 15;
  const int hi = lane >> 4;

  // ---- Q fragments (4 d-chunks of 32), kept in registers ----
  v16h qf[4];
  {
    int qa = q0 + m;
    bool qv = qa < SEQ;
    size_t qrow = ((size_t)(b * SEQ + (qv ? qa : 0)) * NHEAD + hd) * HDIM;
#pragma unroll
    for (int c = 0; c < 4; ++c) {
      HPack16 pk;
      if (qv) {
        const float* p0 = q + qrow + c * 32 + hi * 8;        // k = c*32+hi*8..+7
        const float* p1 = q + qrow + c * 32 + 16 + hi * 8;   // k = +16
        float4 f0 = ((const float4*)p0)[0], f1 = ((const float4*)p0)[1];
        float4 f2 = ((const float4*)p1)[0], f3 = ((const float4*)p1)[1];
        pk.h[0]=(_Float16)f0.x; pk.h[1]=(_Float16)f0.y; pk.h[2]=(_Float16)f0.z; pk.h[3]=(_Float16)f0.w;
        pk.h[4]=(_Float16)f1.x; pk.h[5]=(_Float16)f1.y; pk.h[6]=(_Float16)f1.z; pk.h[7]=(_Float16)f1.w;
        pk.h[8]=(_Float16)f2.x; pk.h[9]=(_Float16)f2.y; pk.h[10]=(_Float16)f2.z; pk.h[11]=(_Float16)f2.w;
        pk.h[12]=(_Float16)f3.x; pk.h[13]=(_Float16)f3.y; pk.h[14]=(_Float16)f3.z; pk.h[15]=(_Float16)f3.w;
      } else {
#pragma unroll
        for (int j = 0; j < 16; ++j) pk.h[j] = (_Float16)0.f;
      }
      qf[c] = pk.v;
    }
  }

  // ---- Phase 1: score tiles round-robin over waves ----
  const float iscale = 0.08838834764831845f;   // 1/sqrt(128)
  for (int kt = w; kt < NKT; kt += 4) {
    v8f accS = {};
    int n    = lane & 15;
    int kabs = kt * 16 + n;
    bool kv  = kabs < SEQ;
    size_t krow = ((size_t)(b * SEQ + (kv ? kabs : 0)) * NHEAD + hd) * HDIM;
#pragma unroll
    for (int c = 0; c < 4; ++c) {
      HPack16 pk;
      if (kv) {
        const float* kp = k + krow + c * 32 + hi * 8;        // d = c*32+hi*8..+7 and +16
        float4 f0 = ((const float4*)kp)[0], f1 = ((const float4*)kp)[1];
        float4 f2 = ((const float4*)kp)[4], f3 = ((const float4*)kp)[5];
        pk.h[0]=(_Float16)f0.x; pk.h[1]=(_Float16)f0.y; pk.h[2]=(_Float16)f0.z; pk.h[3]=(_Float16)f0.w;
        pk.h[4]=(_Float16)f1.x; pk.h[5]=(_Float16)f1.y; pk.h[6]=(_Float16)f1.z; pk.h[7]=(_Float16)f1.w;
        pk.h[8]=(_Float16)f2.x; pk.h[9]=(_Float16)f2.y; pk.h[10]=(_Float16)f2.z; pk.h[11]=(_Float16)f2.w;
        pk.h[12]=(_Float16)f3.x; pk.h[13]=(_Float16)f3.y; pk.h[14]=(_Float16)f3.z; pk.h[15]=(_Float16)f3.w;
      } else {
#pragma unroll
        for (int j = 0; j < 16; ++j) pk.h[j] = (_Float16)0.f;
      }
      accS = wmma_f16(qf[c], pk.v, accS);
    }
    bool mok = kv && (me[b * SEQ + kabs] > 0.f);
#pragma unroll
    for (int r = 0; r < 8; ++r) {
      int mm = r + 8 * hi;
      int qa = q0 + mm;
      bool allowed = mok && (kabs <= qa);
      sc[mm][kt * 16 + n] = allowed ? accS[r] * iscale : -1e9f;
    }
  }
  __syncthreads();

  // ---- Phase 2: softmax, write P in A-frag layout ----
  {
    int row = t >> 3, sub = t & 7;
    float lm = -1e30f;
    for (int c2 = sub; c2 < SKP; c2 += 8) lm = fmaxf(lm, sc[row][c2]);
    red[row][sub] = lm;
    __syncthreads();
    if (sub == 0) {
      float mx = red[row][0];
#pragma unroll
      for (int j = 1; j < 8; ++j) mx = fmaxf(mx, red[row][j]);
      rmax[row] = mx;
    }
    __syncthreads();
    float mx = rmax[row];
    float ls = 0.f;
    for (int c2 = sub; c2 < SKP; c2 += 8) {
      float p = __expf(sc[row][c2] - mx);
      ls += p;
      int ch = c2 >> 5, kk = c2 & 31;
      int pl = row + 16 * ((kk & 15) >= 8);
      int pi = (kk & 7) + 8 * (kk >= 16);
      pf[ch][pl][pi] = (_Float16)p;
    }
    __syncthreads();
    red[row][sub] = ls;
    __syncthreads();
    if (sub == 0) {
      float s = 0.f;
#pragma unroll
      for (int j = 0; j < 8; ++j) s += red[row][j];
      rsum[row] = s;
    }
  }
  __syncthreads();

  // ---- Phase 3: O = P @ V (wave w owns d-tiles 2w, 2w+1) ----
  v8f accO[2] = {};
  for (int ch = 0; ch < NKC; ++ch) {
#pragma unroll
    for (int i = 0; i < 4; ++i) {           // stage V chunk: 512 octet tasks
      int task = t + 128 * i;
      int kk   = task >> 4;
      int oct  = task & 15;
      int ka   = ch * 32 + kk;
      float vv[8];
      if (ka < SEQ) {
        const float* vp = v + ((size_t)(b * SEQ + ka) * NHEAD + hd) * HDIM + oct * 8;
        float4 f0 = ((const float4*)vp)[0], f1 = ((const float4*)vp)[1];
        vv[0]=f0.x; vv[1]=f0.y; vv[2]=f0.z; vv[3]=f0.w;
        vv[4]=f1.x; vv[5]=f1.y; vv[6]=f1.z; vv[7]=f1.w;
      } else {
#pragma unroll
        for (int j = 0; j < 8; ++j) vv[j] = 0.f;
      }
      int dt  = oct >> 1;
      int idx = (kk & 7) + 8 * (kk >> 4);
      int lhi = 16 * ((kk >> 3) & 1);
#pragma unroll
      for (int j = 0; j < 8; ++j) {
        int n2 = (oct & 1) * 8 + j;
        vf[dt][n2 + lhi][idx] = (_Float16)vv[j];
      }
    }
    __syncthreads();
    v16h pa = *(const v16h*)&pf[ch][lane][0];
#pragma unroll
    for (int j = 0; j < 2; ++j) {
      v16h vb = *(const v16h*)&vf[2 * w + j][lane][0];
      accO[j] = wmma_f16(pa, vb, accO[j]);
    }
    __syncthreads();
  }

  // ---- epilogue ----
  const int nn = lane & 15;
#pragma unroll
  for (int j = 0; j < 2; ++j) {
    int dcol = (2 * w + j) * 16 + nn;
#pragma unroll
    for (int r = 0; r < 8; ++r) {
      int mm = r + 8 * hi;
      int qa = q0 + mm;
      if (qa < SEQ)
        o[((size_t)(b * SEQ + qa) * NHEAD + hd) * HDIM + dcol] = accO[j][r] / rsum[mm];
    }
  }
}

// ===========================================================================
// Elementwise / reduction kernels
// ===========================================================================
__global__ __launch_bounds__(256)
void rmsnorm_kernel(const float* __restrict__ x, const float* __restrict__ wgt,
                    float* __restrict__ y, int D)
{
  __shared__ float red[8];
  int row = blockIdx.x;
  const float* xr = x + (size_t)row * D;
  float s = 0.f;
  for (int i = threadIdx.x; i < D; i += 256) { float v = xr[i]; s += v * v; }
  for (int off = 16; off; off >>= 1) s += __shfl_down(s, off);
  if ((threadIdx.x & 31) == 0) red[threadIdx.x >> 5] = s;
  __syncthreads();
  if (threadIdx.x == 0) {
    float tot = 0.f;
#pragma unroll
    for (int j = 0; j < 8; ++j) tot += red[j];
    red[0] = rsqrtf(tot / (float)D + 1e-5f);
  }
  __syncthreads();
  float r = red[0];
  for (int i = threadIdx.x; i < D; i += 256)
    y[(size_t)row * D + i] = xr[i] * r * wgt[i];
}

__global__ __launch_bounds__(256)
void layernorm_kernel(float* __restrict__ x, const float* __restrict__ g,
                      const float* __restrict__ b, int D)
{
  __shared__ float red[8], red2[8];
  int row = blockIdx.x;
  float* xr = x + (size_t)row * D;
  float s = 0.f, ss = 0.f;
  for (int i = threadIdx.x; i < D; i += 256) { float v = xr[i]; s += v; ss += v * v; }
  for (int off = 16; off; off >>= 1) { s += __shfl_down(s, off); ss += __shfl_down(ss, off); }
  if ((threadIdx.x & 31) == 0) { red[threadIdx.x >> 5] = s; red2[threadIdx.x >> 5] = ss; }
  __syncthreads();
  if (threadIdx.x == 0) {
    float t1 = 0.f, t2 = 0.f;
#pragma unroll
    for (int j = 0; j < 8; ++j) { t1 += red[j]; t2 += red2[j]; }
    float mean = t1 / (float)D;
    float var  = t2 / (float)D - mean * mean;
    red[0] = mean; red2[0] = rsqrtf(var + 1e-5f);
  }
  __syncthreads();
  float mean = red[0], rs = red2[0];
  for (int i = threadIdx.x; i < D; i += 256)
    xr[i] = (xr[i] - mean) * rs * g[i] + b[i];
}

__global__ __launch_bounds__(256)
void mean_tokens_kernel(const float* __restrict__ x, float* __restrict__ y, int Tk)
{
  int idx = blockIdx.x * 256 + threadIdx.x;
  if (idx >= BATCH * H_DIM) return;
  int b = idx / H_DIM, c = idx % H_DIM;
  float s = 0.f;
  for (int tt = 0; tt < Tk; ++tt) s += x[((size_t)b * Tk + tt) * H_DIM + c];
  y[idx] = s / (float)Tk;
}

__global__ __launch_bounds__(256)
void build_embeds_kernel(const float* __restrict__ ve, const float* __restrict__ ae,
                         const float* __restrict__ embed, const int* __restrict__ ids,
                         float* __restrict__ X)
{
  size_t idx = (size_t)blockIdx.x * 256 + threadIdx.x;
  if (idx >= (size_t)BATCH * SEQ * H_DIM) return;
  int c = (int)(idx % H_DIM);
  int s = (int)((idx / H_DIM) % SEQ);
  int b = (int)(idx / ((size_t)H_DIM * SEQ));
  float v;
  if (s == 0)      v = ve[b * H_DIM + c];
  else if (s == 1) v = ae[b * H_DIM + c];
  else             v = embed[(size_t)ids[b * T_TOK + (s - 2)] * H_DIM + c];
  X[idx] = v;
}

__global__ __launch_bounds__(256)
void mask_ext_kernel(const float* __restrict__ am, float* __restrict__ me)
{
  int idx = blockIdx.x * 256 + threadIdx.x;
  if (idx >= BATCH * SEQ) return;
  int s = idx % SEQ, b = idx / SEQ;
  me[idx] = (s < 2) ? 1.f : am[b * T_TOK + (s - 2)];
}

__global__ __launch_bounds__(256)
void rope_kernel(float* __restrict__ x)
{
  int idx = blockIdx.x * 256 + threadIdx.x;
  if (idx >= BATCH * SEQ * NHEAD * 64) return;
  int i = idx % 64;
  int h = (idx / 64) % NHEAD;
  int s = (idx / (64 * NHEAD)) % SEQ;
  int b = idx / (64 * NHEAD * SEQ);
  size_t base = ((size_t)(b * SEQ + s) * NHEAD + h) * HDIM;
  float freq = __expf(-(float)i / 64.f * 9.210340371976184f);  // 1/10000^(i/64)
  float ang  = (float)s * freq;
  float c = cosf(ang), sn = sinf(ang);
  float x1 = x[base + i], x2 = x[base + 64 + i];
  x[base + i]      = x1 * c - x2 * sn;
  x[base + 64 + i] = x1 * sn + x2 * c;
}

__global__ __launch_bounds__(256)
void silu_mul_kernel(float* __restrict__ g, const float* __restrict__ u)
{
  size_t idx = (size_t)blockIdx.x * 256 + threadIdx.x;
  if (idx >= (size_t)MROWS * FFN_DIM) return;
  float gv = g[idx];
  g[idx] = gv / (1.f + __expf(-gv)) * u[idx];
}

__global__ __launch_bounds__(256)
void contrast_kernel(const float* __restrict__ ve, const float* __restrict__ ae,
                     const float* __restrict__ embed, const int* __restrict__ ids,
                     const float* __restrict__ temp, float* __restrict__ outc)
{
  __shared__ float acc[14];
  if (threadIdx.x < 14) acc[threadIdx.x] = 0.f;
  __syncthreads();
  const float* tc0 = embed + (size_t)ids[0] * H_DIM;
  const float* tc1 = embed + (size_t)ids[T_TOK] * H_DIM;
  float p[14];
#pragma unroll
  for (int j = 0; j < 14; ++j) p[j] = 0.f;
  for (int kk = threadIdx.x; kk < H_DIM; kk += 256) {
    float v0 = ve[kk], v1 = ve[H_DIM + kk];
    float a0 = ae[kk], a1 = ae[H_DIM + kk];
    float t0 = tc0[kk], t1 = tc1[kk];
    p[0]+=v0*v0; p[1]+=v1*v1; p[2]+=a0*a0; p[3]+=a1*a1; p[4]+=t0*t0; p[5]+=t1*t1;
    p[6]+=v0*t0; p[7]+=v0*t1; p[8]+=v1*t0; p[9]+=v1*t1;
    p[10]+=a0*t0; p[11]+=a0*t1; p[12]+=a1*t0; p[13]+=a1*t1;
  }
#pragma unroll
  for (int j = 0; j < 14; ++j) atomicAdd(&acc[j], p[j]);
  __syncthreads();
  if (threadIdx.x == 0) {
    float nv0 = fmaxf(sqrtf(acc[0]), 1e-12f), nv1 = fmaxf(sqrtf(acc[1]), 1e-12f);
    float na0 = fmaxf(sqrtf(acc[2]), 1e-12f), na1 = fmaxf(sqrtf(acc[3]), 1e-12f);
    float nt0 = fmaxf(sqrtf(acc[4]), 1e-12f), nt1 = fmaxf(sqrtf(acc[5]), 1e-12f);
    float it  = 1.f / temp[0];
    float vs00 = acc[6]  / (nv0 * nt0) * it, vs01 = acc[7]  / (nv0 * nt1) * it;
    float vs10 = acc[8]  / (nv1 * nt0) * it, vs11 = acc[9]  / (nv1 * nt1) * it;
    float as00 = acc[10] / (na0 * nt0) * it, as01 = acc[11] / (na0 * nt1) * it;
    float as10 = acc[12] / (na1 * nt0) * it, as11 = acc[13] / (na1 * nt1) * it;
    auto lse2 = [](float a, float b) {
      float mm = fmaxf(a, b);
      return mm + logf(__expf(a - mm) + __expf(b - mm));
    };
    float cev = -0.5f * ((vs00 - lse2(vs00, vs01)) + (vs11 - lse2(vs10, vs11)));
    float cea = -0.5f * ((as00 - lse2(as00, as01)) + (as11 - lse2(as10, as11)));
    outc[0] = 0.5f * (cev + cea);   // contrast
  }
}

// ---- streaming cross-entropy over LM-head chunks ----
__global__ __launch_bounds__(256)
void ce_init_kernel(float* stmax, float* stsum, float* stlbl)
{
  int r = blockIdx.x * 256 + threadIdx.x;
  if (r < BATCH * (SEQ - 1)) { stmax[r] = -1e30f; stsum[r] = 0.f; stlbl[r] = 0.f; }
}

__global__ __launch_bounds__(256)
void ce_update_kernel(const float* __restrict__ LC, const int* __restrict__ labels,
                      float* __restrict__ stmax, float* __restrict__ stsum,
                      float* __restrict__ stlbl, int chunk_off)
{
  __shared__ float sred[8];
  __shared__ float smx;
  const int r = blockIdx.x;                  // 0 .. B*(SEQ-1)-1
  const int b = r / (SEQ - 1), s = r % (SEQ - 1);
  const float* L = LC + (size_t)(b * SEQ + s) * VC;
  const int t = threadIdx.x;

  float lm = -1e30f;
  for (int i = t; i < VC; i += 256) lm = fmaxf(lm, L[i]);
  for (int off = 16; off; off >>= 1) lm = fmaxf(lm, __shfl_down(lm, off));
  if ((t & 31) == 0) sred[t >> 5] = lm;
  __syncthreads();
  if (t == 0) {
    float mx = sred[0];
#pragma unroll
    for (int j = 1; j < 8; ++j) mx = fmaxf(mx, sred[j]);
    smx = fmaxf(mx, stmax[r]);
  }
  __syncthreads();
  float nm = smx;
  float ps = 0.f;
  for (int i = t; i < VC; i += 256) ps += __expf(L[i] - nm);
  __syncthreads();                            // sred reuse safe after all read smx
  for (int off = 16; off; off >>= 1) ps += __shfl_down(ps, off);
  if ((t & 31) == 0) sred[t >> 5] = ps;
  __syncthreads();
  if (t == 0) {
    float tot = 0.f;
#pragma unroll
    for (int j = 0; j < 8; ++j) tot += sred[j];
    float om = stmax[r];
    stsum[r] = stsum[r] * __expf(om - nm) + tot;
    stmax[r] = nm;
    int lbl = (s >= 1) ? labels[b * T_TOK + (s - 1)] : -100;
    if (lbl >= chunk_off && lbl < chunk_off + VC) stlbl[r] = L[lbl - chunk_off];
  }
}

__global__ __launch_bounds__(256)
void ce_final_kernel(const float* stmax, const float* stsum, const float* stlbl,
                     const float* contrast, float* out)
{
  __shared__ float ssum[256], scnt[256];
  int t = threadIdx.x;
  float s2 = 0.f, c2 = 0.f;
  for (int r = t; r < BATCH * (SEQ - 1); r += 256) {
    int s = r % (SEQ - 1);
    if (s >= 1) {                            // first two ext labels are -100
      s2 += -(stlbl[r] - (logf(stsum[r]) + stmax[r]));
      c2 += 1.f;
    }
  }
  ssum[t] = s2; scnt[t] = c2;
  __syncthreads();
  for (int off = 128; off; off >>= 1) {
    if (t < off) { ssum[t] += ssum[t + off]; scnt[t] += scnt[t + off]; }
    __syncthreads();
  }
  if (t == 0) out[0] = ssum[0] / fmaxf(scnt[0], 1.f) + 0.5f * contrast[0];
}

// ===========================================================================
// Host launch sequence
// ===========================================================================
static inline int cdiv_h(int a, int b) { return (a + b - 1) / b; }

extern "C" void kernel_launch(void* const* d_in, const int* in_sizes, int n_in,
                              void* d_out, int out_size, void* d_ws, size_t ws_size,
                              hipStream_t stream)
{
  (void)in_sizes; (void)n_in; (void)out_size; (void)ws_size;

  const int*   ids    = (const int*)  d_in[0];
  const float* amask  = (const float*)d_in[1];
  const int*   labels = (const int*)  d_in[2];
  const float* vis    = (const float*)d_in[3];
  const float* aco    = (const float*)d_in[4];
  const float* ap_bb  = (const float*)d_in[5];
  const float* ap_b1  = (const float*)d_in[6];
  const float* ap_b2  = (const float*)d_in[7];
  const float* ap_g   = (const float*)d_in[8];
  const float* ap_w1  = (const float*)d_in[9];
  const float* ap_w2  = (const float*)d_in[10];
  const float* embed  = (const float*)d_in[11];
  const float* fnorm  = (const float*)d_in[12];
  const float* lmhead = (const float*)d_in[47];
  const float* temp   = (const float*)d_in[48];
  const float* vp_bb  = (const float*)d_in[49];
  const float* vp_b1  = (const float*)d_in[50];
  const float* vp_b2  = (const float*)d_in[51];
  const float* vp_g   = (const float*)d_in[52];
  const float* vp_w1  = (const float*)d_in[53];
  const float* vp_w2  = (const float*)d_in[54];

  // workspace layout (floats)
  float* ws = (float*)d_ws;
  size_t off = 0;
  auto alloc = [&](size_t n) { float* p = ws + off; off += (n + 63) & ~(size_t)63; return p; };
  float* X   = alloc((size_t)MROWS * H_DIM);
  float* Hn  = alloc((size_t)MROWS * H_DIM);
  float* Qb  = alloc((size_t)MROWS * H_DIM);
  float* Kb  = alloc((size_t)MROWS * H_DIM);
  float* Vb  = alloc((size_t)MROWS * H_DIM);
  float* Ob  = alloc((size_t)MROWS * H_DIM);
  float* Gb  = alloc((size_t)MROWS * FFN_DIM);
  float* Ub  = alloc((size_t)MROWS * FFN_DIM);
  float* Tt  = alloc((size_t)MROWS * RANK);
  float* LC  = alloc((size_t)MROWS * VC);
  float* VE  = alloc((size_t)BATCH * H_DIM);
  float* AE  = alloc((size_t)BATCH * H_DIM);
  float* ME  = alloc((size_t)BATCH * SEQ);
  float* CONTR = alloc(64);
  float* STMAX = alloc(BATCH * (SEQ - 1) + 64);
  float* STSUM = alloc(BATCH * (SEQ - 1) + 64);
  float* STLBL = alloc(BATCH * (SEQ - 1) + 64);

  auto gemm = [&](const float* Xin, const float* Wf, const int* Wq,
                  const float* sc, const float* bias, float* Y,
                  int M2, int N2, int K2, int flags, float alpha) {
    dim3 g(cdiv_h(N2, 128), cdiv_h(M2, 128));
    gemm_kernel<<<g, 256, 0, stream>>>(Xin, Wf, Wq, sc, bias, Y, M2, N2, K2,
                                       flags, alpha);
  };

  // ---- media projectors (reuse Gb/Ub as scratch before layers) ----
  gemm(vis, vp_w1, nullptr, nullptr, vp_b1, Gb, BATCH * 256, 2048, 1024,
       GF_BIAS | GF_RELU, 1.f);
  gemm(Gb, vp_w2, nullptr, nullptr, vp_b2, Ub, BATCH * 256, H_DIM, 2048,
       GF_BIAS, 1.f);
  layernorm_kernel<<<BATCH * 256, 256, 0, stream>>>(Ub, vp_g, vp_bb, H_DIM);
  mean_tokens_kernel<<<cdiv_h(BATCH * H_DIM, 256), 256, 0, stream>>>(Ub, VE, 256);

  gemm(aco, ap_w1, nullptr, nullptr, ap_b1, Gb, BATCH * 512, 1536, 768,
       GF_BIAS | GF_RELU, 1.f);
  gemm(Gb, ap_w2, nullptr, nullptr, ap_b2, Ub, BATCH * 512, H_DIM, 1536,
       GF_BIAS, 1.f);
  layernorm_kernel<<<BATCH * 512, 256, 0, stream>>>(Ub, ap_g, ap_bb, H_DIM);
  mean_tokens_kernel<<<cdiv_h(BATCH * H_DIM, 256), 256, 0, stream>>>(Ub, AE, 512);

  // ---- embeds + masks + contrastive ----
  build_embeds_kernel<<<cdiv_h(BATCH * SEQ * H_DIM, 256), 256, 0, stream>>>(
      VE, AE, embed, ids, X);
  mask_ext_kernel<<<cdiv_h(BATCH * SEQ, 256), 256, 0, stream>>>(amask, ME);
  contrast_kernel<<<1, 256, 0, stream>>>(VE, AE, embed, ids, temp, CONTR);

  // ---- transformer layers ----
  for (int l = 0; l < 2; ++l) {
    int lb = 13 + 17 * l, qb = 55 + 7 * l;
    const float* attn_norm = (const float*)d_in[lb + 0];
    const float* down_s    = (const float*)d_in[lb + 1];
    const float* gate_s    = (const float*)d_in[lb + 2];
    const float* kA        = (const float*)d_in[lb + 3];
    const float* kB        = (const float*)d_in[lb + 4];
    const float* kS        = (const float*)d_in[lb + 5];
    const float* mlp_norm  = (const float*)d_in[lb + 6];
    const float* oA        = (const float*)d_in[lb + 7];
    const float* oB        = (const float*)d_in[lb + 8];
    const float* oS        = (const float*)d_in[lb + 9];
    const float* qA        = (const float*)d_in[lb + 10];
    const float* qB        = (const float*)d_in[lb + 11];
    const float* qS        = (const float*)d_in[lb + 12];
    const float* upS       = (const float*)d_in[lb + 13];
    const float* vA        = (const float*)d_in[lb + 14];
    const float* vB        = (const float*)d_in[lb + 15];
    const float* vS        = (const float*)d_in[lb + 16];
    const int* c_down = (const int*)d_in[qb + 0];
    const int* c_gate = (const int*)d_in[qb + 1];
    const int* c_k    = (const int*)d_in[qb + 2];
    const int* c_o    = (const int*)d_in[qb + 3];
    const int* c_q    = (const int*)d_in[qb + 4];
    const int* c_up   = (const int*)d_in[qb + 5];
    const int* c_v    = (const int*)d_in[qb + 6];

    auto qmm_lora = [&](const float* Xin, const int* codes, const float* sc,
                        const float* A, const float* Bm, float* Y, bool acc) {
      gemm(Xin, nullptr, codes, sc, nullptr, Y, MROWS, H_DIM, H_DIM,
           GF_NF4 | (acc ? GF_ACC : 0), 1.f);
      gemm(Xin, A, nullptr, nullptr, nullptr, Tt, MROWS, RANK, H_DIM, 0, 1.f);
      gemm(Tt, Bm, nullptr, nullptr, nullptr, Y, MROWS, H_DIM, RANK,
           GF_ACC, 2.f);                       // LORA_SCALE fused as alpha
    };

    rmsnorm_kernel<<<MROWS, 256, 0, stream>>>(X, attn_norm, Hn, H_DIM);
    qmm_lora(Hn, c_q, qS, qA, qB, Qb, false);
    qmm_lora(Hn, c_k, kS, kA, kB, Kb, false);
    qmm_lora(Hn, c_v, vS, vA, vB, Vb, false);
    rope_kernel<<<cdiv_h(BATCH * SEQ * NHEAD * 64, 256), 256, 0, stream>>>(Qb);
    rope_kernel<<<cdiv_h(BATCH * SEQ * NHEAD * 64, 256), 256, 0, stream>>>(Kb);
    attn_kernel<<<BATCH * NHEAD * NQT, 128, 0, stream>>>(Qb, Kb, Vb, ME, Ob);
    qmm_lora(Ob, c_o, oS, oA, oB, X, true);    // x = x + o_proj

    rmsnorm_kernel<<<MROWS, 256, 0, stream>>>(X, mlp_norm, Hn, H_DIM);
    gemm(Hn, nullptr, c_gate, gate_s, nullptr, Gb, MROWS, FFN_DIM, H_DIM,
         GF_NF4, 1.f);
    gemm(Hn, nullptr, c_up, upS, nullptr, Ub, MROWS, FFN_DIM, H_DIM,
         GF_NF4, 1.f);
    silu_mul_kernel<<<cdiv_h(MROWS * FFN_DIM, 256), 256, 0, stream>>>(Gb, Ub);
    gemm(Gb, nullptr, c_down, down_s, nullptr, X, MROWS, H_DIM, FFN_DIM,
         GF_NF4 | GF_ACC, 1.f);                // x = x + down
  }

  // ---- final norm + chunked LM-head cross-entropy ----
  rmsnorm_kernel<<<MROWS, 256, 0, stream>>>(X, fnorm, Hn, H_DIM);
  ce_init_kernel<<<cdiv_h(BATCH * (SEQ - 1), 256), 256, 0, stream>>>(STMAX, STSUM, STLBL);
  for (int c = 0; c < NCHUNK; ++c) {
    gemm(Hn, lmhead + (size_t)c * VC * H_DIM, nullptr, nullptr, nullptr, LC,
         MROWS, VC, H_DIM, 0, 1.f);
    ce_update_kernel<<<BATCH * (SEQ - 1), 256, 0, stream>>>(
        LC, labels, STMAX, STSUM, STLBL, c * VC);
  }
  ce_final_kernel<<<1, 256, 0, stream>>>(STMAX, STSUM, STLBL, CONTR, (float*)d_out);
}
